// GCNClassifier_65077344469279
// MI455X (gfx1250) — compile-verified, matrix operand order checked by reading, not run on hardware
//
#include <hip/hip_runtime.h>

// ---------------------------------------------------------------------------
// GCN over dependency trees, CDNA5 (gfx1250) bf16-WMMA implementation.
// Round 2: 2x register blocking per wave + double-buffered LDS (1 barrier/kt).
// ---------------------------------------------------------------------------

typedef __bf16 bf16_t;
typedef __attribute__((ext_vector_type(16))) __bf16 v16bf;
typedef __attribute__((ext_vector_type(8)))  __bf16 v8bf;
typedef __attribute__((ext_vector_type(4)))  __bf16 v4bf;
typedef __attribute__((ext_vector_type(8)))  float  v8f;

static constexpr int Bq   = 128;   // batch
static constexpr int Lq   = 256;   // sequence length
static constexpr int IN   = 360;   // EMB+POSD+NERD
static constexpr int K0   = 384;   // IN padded to mult-of-32
static constexpr int MEMq = 200;   // hidden dim
static constexpr int NP   = 208;   // MEM padded to mult-of-16 (weight rows)
static constexpr int K1   = 224;   // MEM padded to mult-of-32 (layer-2 K)

// ---------------------------------------------------------------------------
// WMMA fragment loader: per-lane, two contiguous 8-element bf16 k-chunks at
// p (k = hi*8..hi*8+7) and p+16 (k = 16+hi*8..). 16B-aligned b128 loads.
// ---------------------------------------------------------------------------
__device__ __forceinline__ v16bf load_frag16(const bf16_t* p) {
  v8bf lo = *(const v8bf*)(p);
  v8bf hh = *(const v8bf*)(p + 16);
  v16bf r;
#pragma unroll
  for (int i = 0; i < 8; ++i) { r[i] = lo[i]; r[i + 8] = hh[i]; }
  return r;
}

// ---------------------------------------------------------------------------
// Kernel 1: embedding gather + concat -> bf16 X0 [B, L, 384] (cols 360.. = 0)
// ---------------------------------------------------------------------------
__global__ void __launch_bounds__(128)
prep_x_kernel(const int* __restrict__ words, const int* __restrict__ pos,
              const int* __restrict__ ner,
              const float* __restrict__ embW, const float* __restrict__ posW,
              const float* __restrict__ nerW, bf16_t* __restrict__ X0) {
  const int bl = blockIdx.x;                 // b*L + l
  const int w = words[bl], p = pos[bl], n = ner[bl];
  const float* er = embW + (size_t)w * 300;
  const float* pr = posW + (size_t)p * 30;
  const float* nr = nerW + (size_t)n * 30;
  bf16_t* dst = X0 + (size_t)bl * K0;
  for (int c = threadIdx.x; c < K0; c += blockDim.x) {
    float v = 0.0f;
    if (c < 300)      v = er[c];
    else if (c < 330) v = pr[c - 300];
    else if (c < 360) v = nr[c - 330];
    dst[c] = (bf16_t)v;
  }
}

// ---------------------------------------------------------------------------
// Kernel 2: pad/convert weights to bf16 [NP x K] row-major, biases to [NP]
// ---------------------------------------------------------------------------
__global__ void __launch_bounds__(256)
prep_w_kernel(const float* __restrict__ W0, const float* __restrict__ b0,
              const float* __restrict__ W1, const float* __restrict__ b1,
              bf16_t* __restrict__ Wp0, bf16_t* __restrict__ Wp1,
              float* __restrict__ bp0, float* __restrict__ bp1) {
  const int idx = blockIdx.x * blockDim.x + threadIdx.x;
  const int stride = gridDim.x * blockDim.x;
  for (int i = idx; i < NP * K0; i += stride) {
    int r = i / K0, c = i % K0;
    float v = (r < MEMq && c < IN) ? W0[r * IN + c] : 0.0f;
    Wp0[i] = (bf16_t)v;
  }
  for (int i = idx; i < NP * K1; i += stride) {
    int r = i / K1, c = i % K1;
    float v = (r < MEMq && c < MEMq) ? W1[r * MEMq + c] : 0.0f;
    Wp1[i] = (bf16_t)v;
  }
  for (int i = idx; i < NP; i += stride) {
    bp0[i] = (i < MEMq) ? b0[i] : 0.0f;
    bp1[i] = (i < MEMq) ? b1[i] : 0.0f;
  }
}

// ---------------------------------------------------------------------------
// Kernel 3: adj f32 -> bf16 rows; rowsum; rdenom = 1/(rowsum+1)
// ---------------------------------------------------------------------------
__global__ void __launch_bounds__(256)
adj_row_kernel(const float* __restrict__ adj, bf16_t* __restrict__ adjh,
               float* __restrict__ rowsum, float* __restrict__ rdenom) {
  const int r = blockIdx.x * blockDim.x + threadIdx.x;   // b*L + i
  if (r >= Bq * Lq) return;
  const float* src = adj + (size_t)r * Lq;
  bf16_t* dst = adjh + (size_t)r * Lq;
  float s = 0.0f;
#pragma unroll 4
  for (int j = 0; j < Lq; j += 4) {
    float4 v = *(const float4*)(src + j);
    s += v.x + v.y + v.z + v.w;
    dst[j]     = (bf16_t)v.x;
    dst[j + 1] = (bf16_t)v.y;
    dst[j + 2] = (bf16_t)v.z;
    dst[j + 3] = (bf16_t)v.w;
  }
  rowsum[r] = s;
  rdenom[r] = 1.0f / (s + 1.0f);
}

// ---------------------------------------------------------------------------
// Kernel 4: colsum[b,j] = sum_i adj[b,i,j]
// ---------------------------------------------------------------------------
__global__ void __launch_bounds__(256)
adj_col_kernel(const float* __restrict__ adj, float* __restrict__ colsum) {
  const int idx = blockIdx.x * blockDim.x + threadIdx.x;  // b*L + j
  if (idx >= Bq * Lq) return;
  const int b = idx / Lq, j = idx % Lq;
  const float* base = adj + ((size_t)b * Lq) * Lq + j;
  float s = 0.0f;
  for (int i = 0; i < Lq; ++i) s += base[(size_t)i * Lq];
  colsum[idx] = s;
}

// ---------------------------------------------------------------------------
// Kernel 5: mask output
// ---------------------------------------------------------------------------
__global__ void __launch_bounds__(256)
mask_kernel(const float* __restrict__ rowsum, const float* __restrict__ colsum,
            float* __restrict__ outMask) {
  const int idx = blockIdx.x * blockDim.x + threadIdx.x;
  if (idx >= Bq * Lq) return;
  outMask[idx] = ((rowsum[idx] + colsum[idx]) == 0.0f) ? 1.0f : 0.0f;
}

// ---------------------------------------------------------------------------
// Kernel 6: message passing  Y = adj @ X + X   (bf16 in / bf16 out, f32 acc)
//   block = 4 waves covering 128 rows x 16 cols; each wave owns a 32x16 tile
//   (two 16x16 accumulators sharing one B fragment). LDS double-buffered.
// ---------------------------------------------------------------------------
__global__ void __launch_bounds__(128)
mp_gemm_kernel(const bf16_t* __restrict__ adjh,   // [B, L, L]
               const bf16_t* __restrict__ X,      // [B, L, ldx]
               bf16_t* __restrict__ Y,            // [B, L, 384]
               int ldx) {
  const int b    = blockIdx.z;
  const int tid  = threadIdx.x;
  const int wave = tid >> 5;
  const int lane = tid & 31;
  const int l15  = lane & 15;
  const int hi   = lane >> 4;
  const int m0   = blockIdx.x * 128 + wave * 32;  // 32 rows per wave
  const int n0   = blockIdx.y * 16;

  const bf16_t* A  = adjh + (size_t)b * Lq * Lq;
  const bf16_t* Bp = X + (size_t)b * Lq * ldx;

  __shared__ bf16_t bs[2][16 * 32];               // transposed [n][k], 2 bufs
  v8f acc0 = {}, acc1 = {};
  const bf16_t* arow0 = A + (size_t)(m0 + l15) * Lq + hi * 8;
  const bf16_t* arow1 = arow0 + (size_t)16 * Lq;

  // staging coords: 4 consecutive n of one k row
  const int sk = (tid * 4) >> 4;                  // 0..31
  const int sn = (tid * 4) & 15;                  // multiple of 4

  int buf = 0;
  for (int kt = 0; kt < Lq; kt += 32, buf ^= 1) {
    {
      v4bf v = *(const v4bf*)(Bp + (size_t)(kt + sk) * ldx + n0 + sn);
#pragma unroll
      for (int j = 0; j < 4; ++j) bs[buf][(sn + j) * 32 + sk] = v[j];
    }
    __syncthreads();
    v16bf a0  = load_frag16(arow0 + kt);
    v16bf a1  = load_frag16(arow1 + kt);
    v16bf bfr = load_frag16(&bs[buf][l15 * 32 + hi * 8]);
    acc0 = __builtin_amdgcn_wmma_f32_16x16x32_bf16(false, a0, false, bfr,
                                                   (short)0, acc0, false, false);
    acc1 = __builtin_amdgcn_wmma_f32_16x16x32_bf16(false, a1, false, bfr,
                                                   (short)0, acc1, false, false);
  }

  const int col = n0 + l15;
#pragma unroll
  for (int v = 0; v < 8; ++v) {                   // residual add + bf16 store
    const int r0 = m0 + hi * 8 + v;
    const int r1 = r0 + 16;
    float y0 = acc0[v] + (float)Bp[(size_t)r0 * ldx + col];
    float y1 = acc1[v] + (float)Bp[(size_t)r1 * ldx + col];
    Y[((size_t)b * Lq + r0) * (size_t)K0 + col] = (bf16_t)y0;
    Y[((size_t)b * Lq + r1) * (size_t)K0 + col] = (bf16_t)y1;
  }
}

// ---------------------------------------------------------------------------
// Kernel 7: dense layer  out = relu((Y @ W^T + 2b) * rdenom)
//   W stored [NP x K] row-major (i.e. B operand is W^T), staged via LDS.
//   Same 32x16-per-wave register blocking + double-buffered LDS.
// ---------------------------------------------------------------------------
template <bool STORE_F32>
__global__ void __launch_bounds__(128)
dense_gemm_kernel(const bf16_t* __restrict__ Yb,   // [B, L, 384]
                  const bf16_t* __restrict__ W,    // [NP, K]
                  const float* __restrict__ bias,  // [NP]
                  const float* __restrict__ rden,  // [B, L]
                  void* __restrict__ outp,         // bf16 [B,L,ldo] or f32 [B,L,200]
                  int K, int ldo) {
  const int b    = blockIdx.z;
  const int tid  = threadIdx.x;
  const int wave = tid >> 5;
  const int lane = tid & 31;
  const int l15  = lane & 15;
  const int hi   = lane >> 4;
  const int m0   = blockIdx.x * 128 + wave * 32;
  const int n0   = blockIdx.y * 16;

  const bf16_t* A = Yb + (size_t)b * Lq * K0;
  __shared__ bf16_t bs[2][16 * 32];
  v8f acc0 = {}, acc1 = {};
  const bf16_t* arow0 = A + (size_t)(m0 + l15) * K0 + hi * 8;
  const bf16_t* arow1 = arow0 + (size_t)16 * K0;

  // staging coords: 4 consecutive k of one weight row n
  const int sk = (tid & 7) * 4;
  const int sn = tid >> 3;

  int buf = 0;
  for (int kt = 0; kt < K; kt += 32, buf ^= 1) {
    {
      v4bf v = *(const v4bf*)(W + (size_t)(n0 + sn) * K + kt + sk);
      *(v4bf*)(&bs[buf][sn * 32 + sk]) = v;
    }
    __syncthreads();
    v16bf a0  = load_frag16(arow0 + kt);
    v16bf a1  = load_frag16(arow1 + kt);
    v16bf bfr = load_frag16(&bs[buf][l15 * 32 + hi * 8]);
    acc0 = __builtin_amdgcn_wmma_f32_16x16x32_bf16(false, a0, false, bfr,
                                                   (short)0, acc0, false, false);
    acc1 = __builtin_amdgcn_wmma_f32_16x16x32_bf16(false, a1, false, bfr,
                                                   (short)0, acc1, false, false);
  }

  const int col = n0 + l15;
  const float b2 = 2.0f * bias[col];
#pragma unroll
  for (int v = 0; v < 8; ++v) {
    const int r0 = m0 + hi * 8 + v;
    const int r1 = r0 + 16;
    float h0 = fmaxf((acc0[v] + b2) * rden[b * Lq + r0], 0.0f);
    float h1 = fmaxf((acc1[v] + b2) * rden[b * Lq + r1], 0.0f);
    if (STORE_F32) {
      if (col < MEMq) {
        ((float*)outp)[((size_t)b * Lq + r0) * (size_t)MEMq + col] = h0;
        ((float*)outp)[((size_t)b * Lq + r1) * (size_t)MEMq + col] = h1;
      }
    } else {
      ((bf16_t*)outp)[((size_t)b * Lq + r0) * (size_t)ldo + col] = (bf16_t)h0;
      ((bf16_t*)outp)[((size_t)b * Lq + r1) * (size_t)ldo + col] = (bf16_t)h1;
    }
  }
}

// ---------------------------------------------------------------------------
// Host-side launch
// ---------------------------------------------------------------------------
extern "C" void kernel_launch(void* const* d_in, const int* in_sizes, int n_in,
                              void* d_out, int out_size, void* d_ws, size_t ws_size,
                              hipStream_t stream) {
  (void)in_sizes; (void)n_in; (void)out_size; (void)ws_size;
  const int*   words = (const int*)d_in[0];
  const int*   pos   = (const int*)d_in[1];
  const int*   ner   = (const int*)d_in[2];
  const float* adj   = (const float*)d_in[3];
  const float* embW  = (const float*)d_in[4];
  const float* posW  = (const float*)d_in[5];
  const float* nerW  = (const float*)d_in[6];
  const float* W0    = (const float*)d_in[7];
  const float* b0    = (const float*)d_in[8];
  const float* W1    = (const float*)d_in[9];
  const float* b1    = (const float*)d_in[10];

  // workspace carve-up (256B aligned)
  size_t off = 0;
  auto carve = [&](size_t bytes) -> void* {
    void* p = (char*)d_ws + off;
    off += (bytes + 255) & ~(size_t)255;
    return p;
  };
  bf16_t* adjh   = (bf16_t*)carve((size_t)Bq * Lq * Lq * sizeof(bf16_t));
  bf16_t* X0     = (bf16_t*)carve((size_t)Bq * Lq * K0 * sizeof(bf16_t));
  bf16_t* Yb     = (bf16_t*)carve((size_t)Bq * Lq * K0 * sizeof(bf16_t));
  bf16_t* X1     = (bf16_t*)carve((size_t)Bq * Lq * K1 * sizeof(bf16_t));
  float*  rowsum = (float*)carve((size_t)Bq * Lq * sizeof(float));
  float*  colsum = (float*)carve((size_t)Bq * Lq * sizeof(float));
  float*  rdenom = (float*)carve((size_t)Bq * Lq * sizeof(float));
  bf16_t* Wp0    = (bf16_t*)carve((size_t)NP * K0 * sizeof(bf16_t));
  bf16_t* Wp1    = (bf16_t*)carve((size_t)NP * K1 * sizeof(bf16_t));
  float*  bp0    = (float*)carve((size_t)NP * sizeof(float));
  float*  bp1    = (float*)carve((size_t)NP * sizeof(float));

  float* out_x    = (float*)d_out;                                  // [B,L,200]
  float* out_mask = (float*)d_out + (size_t)Bq * Lq * MEMq;         // [B,L,1]

  // X1 pad columns (208..223) must be zero for the layer-2 K loop
  hipMemsetAsync(X1, 0, (size_t)Bq * Lq * K1 * sizeof(bf16_t), stream);

  // prep
  prep_x_kernel<<<Bq * Lq, 128, 0, stream>>>(words, pos, ner, embW, posW, nerW, X0);
  prep_w_kernel<<<208, 256, 0, stream>>>(W0, b0, W1, b1, Wp0, Wp1, bp0, bp1);
  adj_row_kernel<<<(Bq * Lq) / 256, 256, 0, stream>>>(adj, adjh, rowsum, rdenom);
  adj_col_kernel<<<(Bq * Lq) / 256, 256, 0, stream>>>(adj, colsum);
  mask_kernel<<<(Bq * Lq) / 256, 256, 0, stream>>>(rowsum, colsum, out_mask);

  // layer 1: Y = adj@X0 + X0 ; X1 = relu((Y@W0^T + 2b0) * rdenom)
  mp_gemm_kernel<<<dim3(Lq / 128, K0 / 16, Bq), 128, 0, stream>>>(adjh, X0, Yb, K0);
  dense_gemm_kernel<false><<<dim3(Lq / 128, NP / 16, Bq), 128, 0, stream>>>(
      Yb, Wp0, bp0, rdenom, X1, K0, K1);

  // layer 2: Y = adj@X1 + X1 ; out = relu((Y@W1^T + 2b1) * rdenom)
  mp_gemm_kernel<<<dim3(Lq / 128, K1 / 16, Bq), 128, 0, stream>>>(adjh, X1, Yb, K1);
  dense_gemm_kernel<true><<<dim3(Lq / 128, NP / 16, Bq), 128, 0, stream>>>(
      Yb, Wp1, bp1, rdenom, out_x, K1, MEMq);
}
